// MinLSTMLM_32916629356934
// MI455X (gfx1250) — matrix operand order repaired and checked by reading, not compiled, not run on hardware
//
#include <hip/hip_runtime.h>
#include <hip/hip_bf16.h>
#include <cstdint>

// ---------------- model constants ----------------
#define VOCAB 32000
#define HDIM  1024
#define NLAYR 6
#define IDIM  1536
#define FDIM  4096
#define BATCH 2
#define SEQ   1024
#define BS    (BATCH * SEQ)            // 2048 rows
#define EPSV  1e-8f
#define LNEPS 1e-5f
#define LOG_EPSV (-18.420680743952367f) // log(1e-8)

// ---------------- vector types ----------------
typedef __attribute__((ext_vector_type(16))) __bf16 bf16x16;
typedef __attribute__((ext_vector_type(8)))  float  floatx8;
typedef __attribute__((ext_vector_type(4)))  unsigned int u32x4;
typedef __attribute__((ext_vector_type(8)))  int          i32x8;
typedef __attribute__((ext_vector_type(4)))  int          i32x4;

union BF16Frag { bf16x16 v; uint4 q[2]; unsigned int u[8]; };
union AccFrag  { floatx8 v; float f[8]; };

__device__ inline floatx8 wmma_bf16(bf16x16 a, bf16x16 b, floatx8 c) {
#if defined(__HIP_DEVICE_COMPILE__)
  return __builtin_amdgcn_wmma_f32_16x16x32_bf16(
      false, a, false, b, (short)0, c, false, false);
#else
  (void)a; (void)b;
  return c;
#endif
}

__device__ inline unsigned short f2bf(float f) {
  unsigned int u = __float_as_uint(f);
  unsigned int r = u + 0x7FFFu + ((u >> 16) & 1u);
  return (unsigned short)(r >> 16);
}

__device__ inline float logaddexpf_dev(float a, float b) {
  float m = fmaxf(a, b), n = fminf(a, b);
  return m + log1pf(expf(n - m));
}

// low 32 bits of a generic pointer to __shared__ = LDS byte offset (aperture map)
__device__ inline unsigned int lds_byte_off(const void* p) {
  return (unsigned int)(unsigned long long)p;
}

// ---------------- Tensor Data Mover: 2-D tile (rows x 32 bf16) global -> LDS ----------
// Row length fixed at 32 elements (64 B = 16 dwords); LDS padding of 4 dwords (16 B)
// after every row -> 80 B LDS row stride, keeps b128 alignment and rotates banks.
__device__ inline void tdm_load_tile32(const unsigned short* gptr,
                                       unsigned int lds_off,
                                       int K_elems, int tensor_rows, int tile_rows) {
#if defined(__HIP_DEVICE_COMPILE__)
#if __has_builtin(__builtin_amdgcn_tensor_load_to_lds)
  unsigned long long ga = (unsigned long long)gptr;
  u32x4 g0;
  g0.x = 1u;                                            // count=1, user mode
  g0.y = lds_off;                                       // lds_addr
  g0.z = (unsigned int)(ga & 0xFFFFFFFFu);              // global_addr[31:0]
  g0.w = (unsigned int)((ga >> 32) & 0x1FFFFFFu)        // global_addr[56:32]
       | (2u << 30);                                    // type = 2 (image/D#)
  unsigned int td0 = (unsigned int)K_elems;             // tensor_dim0
  unsigned int td1 = (unsigned int)tensor_rows;         // tensor_dim1
  unsigned long long st0 = (unsigned long long)K_elems; // tensor_dim0_stride
  i32x8 g1;
  g1[0] = (int)((1u << 16)      // data_size = 2 bytes
              | (1u << 20)      // pad_enable
              | (3u << 22)      // pad_interval: 16 dwords
              | (3u << 25));    // pad_amount: 4 dwords
  g1[1] = (int)((td0 & 0xFFFFu) << 16);
  g1[2] = (int)(((td0 >> 16) & 0xFFFFu) | ((td1 & 0xFFFFu) << 16));
  g1[3] = (int)(((td1 >> 16) & 0xFFFFu) | (32u << 16));            // tile_dim0 = 32
  g1[4] = (int)((unsigned int)tile_rows & 0xFFFFu);                // tile_dim1
  g1[5] = (int)(unsigned int)(st0 & 0xFFFFFFFFu);
  g1[6] = (int)(unsigned int)((st0 >> 32) & 0xFFFFu);
  g1[7] = 0;
  i32x4 z4 = {0, 0, 0, 0};
  i32x8 z8 = {0, 0, 0, 0, 0, 0, 0, 0};
  // clang-23 / amdgpu-toolchain 6-arg form: (g0, g1, g2, g3, g4, cpol)
  __builtin_amdgcn_tensor_load_to_lds(g0, g1, z4, z4, z8, 0);
#endif
#else
  (void)gptr; (void)lds_off; (void)K_elems; (void)tensor_rows; (void)tile_rows;
#endif
}

__device__ inline void wait_tensor0() {
#if defined(__HIP_DEVICE_COMPILE__)
#if __has_builtin(__builtin_amdgcn_s_wait_tensorcnt)
  __builtin_amdgcn_s_wait_tensorcnt(0);
#endif
#endif
}

// ---------------- embedding gather ----------------
__global__ void embed_kernel(const int* __restrict__ tokens,
                             const float* __restrict__ emb,
                             float* __restrict__ x) {
  int row = blockIdx.x;
  int t = tokens[row];
  const float* src = emb + (size_t)t * HDIM;
  float* dst = x + (size_t)row * HDIM;
  for (int i = threadIdx.x; i < HDIM; i += blockDim.x) dst[i] = src[i];
}

// ---------------- layernorm (fp32 in -> bf16 out) ----------------
__global__ __launch_bounds__(256)
void ln_kernel(const float* __restrict__ x,
               const float* __restrict__ g,
               const float* __restrict__ b,
               unsigned short* __restrict__ out) {
  int row = blockIdx.x;
  const float* xr = x + (size_t)row * HDIM;
  float s = 0.f, s2 = 0.f;
  for (int i = threadIdx.x; i < HDIM; i += blockDim.x) {
    float v = xr[i]; s += v; s2 += v * v;
  }
  #pragma unroll
  for (int off = 16; off > 0; off >>= 1) {
    s  += __shfl_xor(s,  off, 32);
    s2 += __shfl_xor(s2, off, 32);
  }
  __shared__ float red0[8], red1[8];
  int wave = threadIdx.x >> 5, lane = threadIdx.x & 31;
  if (lane == 0) { red0[wave] = s; red1[wave] = s2; }
  __syncthreads();
  if (threadIdx.x == 0) {
    float a = 0.f, c = 0.f;
    for (int w = 0; w < 8; ++w) { a += red0[w]; c += red1[w]; }
    red0[0] = a; red1[0] = c;
  }
  __syncthreads();
  s = red0[0]; s2 = red1[0];
  float mean = s * (1.0f / HDIM);
  float var  = s2 * (1.0f / HDIM) - mean * mean;
  float inv  = rsqrtf(var + LNEPS);
  unsigned short* o = out + (size_t)row * HDIM;
  for (int i = threadIdx.x; i < HDIM; i += blockDim.x) {
    float v = (xr[i] - mean) * inv * g[i] + b[i];
    o[i] = f2bf(v);
  }
}

// ---------------- weight convert + transpose: W[K,N] f32 -> Wt[N,K] bf16 ----------
__global__ __launch_bounds__(256)
void convert_transpose_kernel(const float* __restrict__ W,
                              unsigned short* __restrict__ Wt,
                              int K, int N) {
  __shared__ float tile[32][33];
  int n0 = blockIdx.x * 32;
  int k0 = blockIdx.y * 32;
  int tx = threadIdx.x;
  int ty = threadIdx.y;
  #pragma unroll
  for (int i = ty; i < 32; i += 8)
    tile[i][tx] = W[(size_t)(k0 + i) * N + (n0 + tx)];
  __syncthreads();
  #pragma unroll
  for (int i = ty; i < 32; i += 8)
    Wt[(size_t)(n0 + i) * K + (k0 + tx)] = f2bf(tile[tx][i]);
}

// ---------------- minLSTM gate math + sequential scan over S ----------------
__global__ void gate_scan_kernel(const float* __restrict__ gh,
                                 unsigned short* __restrict__ cellbf) {
  int idx = blockIdx.x * blockDim.x + threadIdx.x;  // b*IDIM + i
  if (idx >= BATCH * IDIM) return;
  int b = idx / IDIM;
  int i = idx - b * IDIM;
  const float* base = gh + (size_t)b * SEQ * (3 * IDIM);
  float cum = 0.f;
  for (int s = 0; s < SEQ; ++s) {
    const float* r = base + (size_t)s * (3 * IDIM);
    if (s + 1 < SEQ) __builtin_prefetch(r + 3 * IDIM + i, 0, 1);  // global_prefetch_b8
    float fg = r[i];
    float ig = r[IDIM + i];
    float th = r[2 * IDIM + i];
    float sf = 1.0f / (1.0f + expf(-fg));
    float si = 1.0f / (1.0f + expf(-ig));
    float log_f = logf(sf + EPSV);
    float log_i = logf(si + EPSV);
    float lse   = logaddexpf_dev(log_f, log_i);
    float log_fp = lse - log_f;          // source quirk: lse - log_coeffs
    float log_ip = lse - log_i;
    float log_state = logaddexpf_dev(log_fp + LOG_EPSV, log_ip + logf(th + EPSV));
    cum += log_fp;
    float cell = expf(cum + log_state);
    cellbf[((size_t)b * SEQ + s) * IDIM + i] = f2bf(cell);
  }
}

// ---------------- TDM double-buffered bf16 WMMA GEMM ----------------
// C[M,N] = A[M,K](bf16, row-major) * Wt[N,K](bf16, N-major)
// block = 256 threads = 8 waves (2 M-waves x 4 N-waves); block tile 32x256
// per K-chunk (32): wave0 DMAs A(32x32) and B(256x32) tiles into LDS via TDM,
// double-buffered; all waves consume via ds_load_b128 and run 4 WMMAs each.
// EPI: 0 store f32; 1 Cf += acc; 2 bf16(gelu(acc+bias)); 3 Cf += acc + bias
template <int EPI>
__global__ __launch_bounds__(256)
void gemm_bf16_tdm_kernel(const unsigned short* __restrict__ A,
                          const unsigned short* __restrict__ Bt,
                          const float* __restrict__ bias,
                          float* __restrict__ Cf,
                          unsigned short* __restrict__ Cbf,
                          int M, int N, int K) {
  // 80B row stride (32 elems + 16B TDM pad) => bank rotation + b128 alignment
  __shared__ unsigned short Asm[2][32][40];    //  5.0 KB
  __shared__ unsigned short Bsm[2][256][40];   // 40.0 KB
  const int lane  = threadIdx.x & 31;
  const int wave  = threadIdx.x >> 5;
  const int wy    = wave >> 2;               // 0..1
  const int wx    = wave & 3;                // 0..3
  const int m0    = blockIdx.y * 32;
  const int n0    = blockIdx.x * 256;
  const int nlane = lane & 15;
  const int h     = lane >> 4;

  const unsigned short* Ab = A  + (size_t)m0 * K;
  const unsigned short* Bb = Bt + (size_t)n0 * K;

  AccFrag acc[4];
  #pragma unroll
  for (int t = 0; t < 4; ++t)
    #pragma unroll
    for (int r = 0; r < 8; ++r) acc[t].f[r] = 0.f;

  const int nc = K >> 5;
  if (wave == 0) {
    tdm_load_tile32(Ab, lds_byte_off(&Asm[0][0][0]), K, M, 32);
    tdm_load_tile32(Bb, lds_byte_off(&Bsm[0][0][0]), K, N, 256);
  }

  for (int c = 0; c < nc; ++c) {
    const int cur = c & 1;
    if (wave == 0) wait_tensor0();     // chunk c resident in LDS
    __syncthreads();                   // publish to all waves; prior compute done
    if (wave == 0 && (c + 1) < nc) {   // overlap chunk c+1 DMA with compute
      tdm_load_tile32(Ab + (c + 1) * 32, lds_byte_off(&Asm[cur ^ 1][0][0]), K, M, 32);
      tdm_load_tile32(Bb + (c + 1) * 32, lds_byte_off(&Bsm[cur ^ 1][0][0]), K, N, 256);
    }
    // A fragment (16-bit A 16x32 layout): K = 8h..8h+7, then 16+8h..16+8h+7
    BF16Frag af;
    const unsigned short* ap = &Asm[cur][wy * 16 + nlane][8 * h];
    af.q[0] = *reinterpret_cast<const uint4*>(ap);
    af.q[1] = *reinterpret_cast<const uint4*>(ap + 16);
    #pragma unroll
    for (int t = 0; t < 4; ++t) {
      // B fragment (B 32x16 layout): lane n holds K = 8h..8h+7, then 16+8h..16+8h+7
      BF16Frag bf;
      const unsigned short* bp = &Bsm[cur][wx * 64 + t * 16 + nlane][8 * h];
      bf.q[0] = *reinterpret_cast<const uint4*>(bp);
      bf.q[1] = *reinterpret_cast<const uint4*>(bp + 16);
      acc[t].v = wmma_bf16(af.v, bf.v, acc[t].v);
    }
  }

  // epilogue: C layout — lane 0..15 = N, VGPR r -> M = r + 8*(lane>>4)
  const int mbase = m0 + wy * 16 + h * 8;
  #pragma unroll
  for (int t = 0; t < 4; ++t) {
    const int col = n0 + wx * 64 + t * 16 + nlane;
    #pragma unroll
    for (int r = 0; r < 8; ++r) {
      const int row = mbase + r;
      const size_t idx = (size_t)row * N + col;
      const float v = acc[t].f[r];
      if (EPI == 0) {
        Cf[idx] = v;
      } else if (EPI == 1) {
        Cf[idx] = Cf[idx] + v;
      } else if (EPI == 2) {
        float gsum = v + bias[col];
        float gl = 0.5f * gsum * (1.0f + erff(gsum * 0.70710678118654752f));
        Cbf[idx] = f2bf(gl);
      } else {
        Cf[idx] = Cf[idx] + v + bias[col];
      }
    }
  }
}

// ---------------- host-side orchestration ----------------
static inline size_t alignup(size_t v) { return (v + 255) & ~(size_t)255; }

static void launch_gemm(int epi,
                        const unsigned short* A, const unsigned short* Bt,
                        const float* bias, float* Cf, unsigned short* Cbf,
                        int M, int N, int K, hipStream_t stream) {
  dim3 grid(N / 256, M / 32), block(256);
  switch (epi) {
    case 0: gemm_bf16_tdm_kernel<0><<<grid, block, 0, stream>>>(A, Bt, bias, Cf, Cbf, M, N, K); break;
    case 1: gemm_bf16_tdm_kernel<1><<<grid, block, 0, stream>>>(A, Bt, bias, Cf, Cbf, M, N, K); break;
    case 2: gemm_bf16_tdm_kernel<2><<<grid, block, 0, stream>>>(A, Bt, bias, Cf, Cbf, M, N, K); break;
    default: gemm_bf16_tdm_kernel<3><<<grid, block, 0, stream>>>(A, Bt, bias, Cf, Cbf, M, N, K); break;
  }
}

static void launch_convT(const float* W, unsigned short* Wt, int K, int N,
                         hipStream_t stream) {
  dim3 grid(N / 32, K / 32), block(32, 8);
  convert_transpose_kernel<<<grid, block, 0, stream>>>(W, Wt, K, N);
}

extern "C" void kernel_launch(void* const* d_in, const int* in_sizes, int n_in,
                              void* d_out, int out_size, void* d_ws, size_t ws_size,
                              hipStream_t stream) {
  (void)in_sizes; (void)n_in; (void)out_size; (void)ws_size;

  const int*   tokens  = (const int*)  d_in[0];
  const float* emb     = (const float*)d_in[1];
  const float* ln1_g   = (const float*)d_in[2];
  const float* ln1_b   = (const float*)d_in[3];
  const float* w_gh    = (const float*)d_in[4];
  const float* w_out   = (const float*)d_in[5];
  const float* ln2_g   = (const float*)d_in[6];
  const float* ln2_b   = (const float*)d_in[7];
  const float* w1      = (const float*)d_in[8];
  const float* b1      = (const float*)d_in[9];
  const float* w2      = (const float*)d_in[10];
  const float* b2      = (const float*)d_in[11];
  const float* lnf_g   = (const float*)d_in[12];
  const float* lnf_b   = (const float*)d_in[13];
  const float* w_logit = (const float*)d_in[14];
  float* logits = (float*)d_out;

  char* ws = (char*)d_ws;
  float*          x      = (float*)ws;          ws += alignup((size_t)BS * HDIM * 4);
  unsigned short* hbf    = (unsigned short*)ws; ws += alignup((size_t)BS * HDIM * 2);
  float*          gh     = (float*)ws;          ws += alignup((size_t)BS * 3 * IDIM * 4);
  unsigned short* cellbf = (unsigned short*)ws; ws += alignup((size_t)BS * IDIM * 2);
  unsigned short* ffnbf  = (unsigned short*)ws; ws += alignup((size_t)BS * FDIM * 2);
  unsigned short* wt     = (unsigned short*)ws; ws += alignup((size_t)HDIM * VOCAB * 2);

  embed_kernel<<<dim3(BS), dim3(256), 0, stream>>>(tokens, emb, x);

  for (int l = 0; l < NLAYR; ++l) {
    // --- minLSTM branch ---
    ln_kernel<<<dim3(BS), dim3(256), 0, stream>>>(x, ln1_g + l * HDIM, ln1_b + l * HDIM, hbf);
    launch_convT(w_gh + (size_t)l * HDIM * 3 * IDIM, wt, HDIM, 3 * IDIM, stream);
    launch_gemm(0, hbf, wt, nullptr, gh, nullptr, BS, 3 * IDIM, HDIM, stream);
    gate_scan_kernel<<<dim3((BATCH * IDIM + 255) / 256), dim3(256), 0, stream>>>(gh, cellbf);
    launch_convT(w_out + (size_t)l * IDIM * HDIM, wt, IDIM, HDIM, stream);
    launch_gemm(1, cellbf, wt, nullptr, x, nullptr, BS, HDIM, IDIM, stream);

    // --- FFN branch ---
    ln_kernel<<<dim3(BS), dim3(256), 0, stream>>>(x, ln2_g + l * HDIM, ln2_b + l * HDIM, hbf);
    launch_convT(w1 + (size_t)l * HDIM * FDIM, wt, HDIM, FDIM, stream);
    launch_gemm(2, hbf, wt, b1 + l * FDIM, nullptr, ffnbf, BS, FDIM, HDIM, stream);
    launch_convT(w2 + (size_t)l * FDIM * HDIM, wt, FDIM, HDIM, stream);
    launch_gemm(3, ffnbf, wt, b2 + l * HDIM, x, nullptr, BS, HDIM, FDIM, stream);
  }

  ln_kernel<<<dim3(BS), dim3(256), 0, stream>>>(x, lnf_g, lnf_b, hbf);
  launch_convT(w_logit, wt, HDIM, VOCAB, stream);
  launch_gemm(0, hbf, wt, nullptr, logits, nullptr, BS, VOCAB, HDIM, stream);
}
